// GATLayer_73753178406980
// MI455X (gfx1250) — compile-verified, matrix operand order checked by reading, not run on hardware
//
#include <hip/hip_runtime.h>
#include <hip/hip_bf16.h>

// CDNA5 / gfx1250: wave32, WMMA 16x16x32 bf16 (f32 accumulate), TDM staging.
typedef __attribute__((ext_vector_type(16))) __bf16       v16bf;
typedef __attribute__((ext_vector_type(8)))  float        v8f;
typedef __attribute__((ext_vector_type(4)))  unsigned int u32x4;
typedef __attribute__((ext_vector_type(8)))  int          i32x8;
typedef __attribute__((ext_vector_type(4)))  int          i32x4;

#define DD 128          // feature dim
#define HH 512          // hidden dim
#define LRELU_SLOPE 0.01f
#define LN_EPS 1e-5f

static __device__ __forceinline__ v8f wmma_bf16(v16bf a, v16bf b, v8f c) {
  // 8 args: (neg_a, A, neg_b, B, c_mod, C, reuse_a, reuse_b)
  return __builtin_amdgcn_wmma_f32_16x16x32_bf16(false, a, false, b, (short)0, c,
                                                 false, false);
}

// Load 16 consecutive f32 and convert to a bf16 fragment slice (A side only).
static __device__ __forceinline__ v16bf cvt16(const float* __restrict__ p) {
  v16bf r;
#pragma unroll
  for (int i = 0; i < 16; ++i) r[i] = (__bf16)p[i];
  return r;
}

// ---------------------------------------------------------------------------
// Prep: f32 -> bf16 copy (weights converted once per call; they then serve
// every block as 32B fragments with zero inner-loop conversion VALU).
// ---------------------------------------------------------------------------
__global__ void k_cvt(const float* __restrict__ in, __bf16* __restrict__ out, int n) {
  const int i = blockIdx.x * 256 + threadIdx.x;
  if (i < n) out[i] = (__bf16)in[i];
}

// ---------------------------------------------------------------------------
// Kernel 1: Z = S @ Wfc^T  via bf16 WMMA (f32 accumulate, f32 out).
// One wave per 16-row M tile; wave computes all 128 output columns.
//   A frag: lane L reads S row (L%16), K = k0 + (L<16?0:16) .. +15 (contig, cvt)
//   B frag: lane L reads Wb row (L%16 of the 16-col group), same K slice (bf16)
// ---------------------------------------------------------------------------
__global__ void k_gemm_z(const float* __restrict__ S, const __bf16* __restrict__ WB,
                         float* __restrict__ Z, int Mtiles) {
  const int lane  = threadIdx.x & 31;
  const int wv    = threadIdx.x >> 5;
  const int mtile = blockIdx.x * 8 + wv;
  if (mtile >= Mtiles) return;                 // wave-uniform: EXEC stays all-1s
  const int mbase  = mtile * 16;
  const int lrow16 = lane & 15;
  const int khalf  = (lane < 16) ? 0 : 16;
  const int mofs   = (lane < 16) ? 0 : 8;

  v8f acc[8] = {};
#pragma unroll
  for (int k0 = 0; k0 < DD; k0 += 32) {
    v16bf afrag = cvt16(S + (size_t)(mbase + lrow16) * DD + k0 + khalf);
#pragma unroll
    for (int n = 0; n < 8; ++n) {
      v16bf bfrag = *(const v16bf*)(WB + (size_t)(n * 16 + lrow16) * DD + k0 + khalf);
      acc[n] = wmma_bf16(afrag, bfrag, acc[n]);
    }
  }
#pragma unroll
  for (int n = 0; n < 8; ++n)
#pragma unroll
    for (int r = 0; r < 8; ++r)
      Z[(size_t)(mbase + r + mofs) * DD + n * 16 + lrow16] = acc[n][r];
}

// ---------------------------------------------------------------------------
// Kernel 2: el[i] = leaky_relu( dot(Z[i,:], a[:128]) ) — one wave per row.
// ---------------------------------------------------------------------------
__global__ void k_row_dot(const float* __restrict__ Z, const float* __restrict__ A,
                          float* __restrict__ EL, int N) {
  const int wave = blockIdx.x * 8 + (threadIdx.x >> 5);
  const int lane = threadIdx.x & 31;
  if (wave >= N) return;
  const float* zp = Z + (size_t)wave * DD;
  float s = 0.f;
#pragma unroll
  for (int i = 0; i < 4; ++i) s += zp[lane + 32 * i] * A[lane + 32 * i];
#pragma unroll
  for (int off = 16; off > 0; off >>= 1) s += __shfl_xor(s, off, 32);
  if (lane == 0) EL[wave] = (s > 0.f) ? s : LRELU_SLOPE * s;
}

// ---------------------------------------------------------------------------
// Kernel 3: segment softmax over sorted dst + aggregation + elu + residual c.
// One 128-thread block per destination node (edges contiguous; binary search).
// Emits hc in f32 (residual) and bf16 (TDM-stageable copy for the FFN).
// ---------------------------------------------------------------------------
__global__ void k_gat_agg(const float* __restrict__ EL, const float* __restrict__ Z,
                          const int* __restrict__ SRC, const int* __restrict__ DST,
                          const float* __restrict__ C, float* __restrict__ HC,
                          __bf16* __restrict__ HCB, int E) {
  const int node = blockIdx.x;
  const int tid  = threadIdx.x;
  __shared__ int   bounds[2];
  __shared__ float rm[128], rs[128];
  __shared__ float salpha[128];
  __shared__ int   ssrc[128];

  if (tid == 0) {
    int lo = 0, hi = E;
    while (lo < hi) { int mid = (lo + hi) >> 1; if (DST[mid] < node) lo = mid + 1; else hi = mid; }
    bounds[0] = lo;
    hi = E;
    while (lo < hi) { int mid = (lo + hi) >> 1; if (DST[mid] <= node) lo = mid + 1; else hi = mid; }
    bounds[1] = lo;
  }
  __syncthreads();
  const int e0 = bounds[0], e1 = bounds[1];

  // online softmax statistics (finite sentinel avoids inf-inf NaNs)
  float m = -1e30f, ssum = 0.f;
  for (int j = e0 + tid; j < e1; j += 128) {
    float e = EL[SRC[j]];
    if (e > m) { ssum *= __expf(m - e); m = e; }
    ssum += __expf(e - m);
  }
  rm[tid] = m; rs[tid] = ssum;
  __syncthreads();
#pragma unroll
  for (int off = 64; off > 0; off >>= 1) {
    if (tid < off) {
      float m1 = rm[tid], s1 = rs[tid];
      float m2 = rm[tid + off], s2 = rs[tid + off];
      float M = fmaxf(m1, m2);
      rm[tid] = M;
      rs[tid] = s1 * __expf(m1 - M) + s2 * __expf(m2 - M);
    }
    __syncthreads();
  }
  const float Mall = rm[0];
  const float Sinv = (rs[0] > 0.f) ? 1.f / rs[0] : 0.f;

  // column-parallel weighted aggregation; thread tid owns column tid.
  float acc = 0.f;
  for (int base = e0; base < e1; base += 128) {
    const int j = base + tid;
    if (j < e1) {
      const int sidx = SRC[j];
      ssrc[tid]   = sidx;
      salpha[tid] = __expf(EL[sidx] - Mall) * Sinv;
    }
    __syncthreads();
    const int cnt = min(128, e1 - base);
    for (int k = 0; k < cnt; ++k)
      acc += salpha[k] * Z[(size_t)ssrc[k] * DD + tid];  // z resident in 192MB L2
    __syncthreads();
  }
  const float h   = acc;
  const float el1 = (h > 0.f) ? h : (__expf(h) - 1.f);   // elu
  const float out = el1 + C[(size_t)node * DD + tid];
  HC [(size_t)node * DD + tid] = out;
  HCB[(size_t)node * DD + tid] = (__bf16)out;
}

// ---------------------------------------------------------------------------
// Kernel 4: fused FFN. y = relu(hc@w1^T + b1)@w2^T + b2 + hc (pre-LN).
// Block = 256 threads = 8 waves = 128 rows. The bf16 hc tile is DMA'd into
// LDS by the Tensor Data Mover (D# descriptor, TENSORcnt), then two WMMA
// GEMMs per 128-wide H chunk with y accumulated in registers.
// ---------------------------------------------------------------------------
__global__ void k_ffn(const float* __restrict__ HC, const __bf16* __restrict__ HCB,
                      const __bf16* __restrict__ W1B, const float* __restrict__ B1,
                      const __bf16* __restrict__ W2B, const float* __restrict__ B2,
                      float* __restrict__ Y, int Nc, int Mtiles) {
  __shared__ __bf16 sHC[128 * DD];   // 32 KB, written by TDM
  __shared__ __bf16 sT[128 * 128];   // 32 KB
  const int tid   = threadIdx.x;
  const int lane  = tid & 31;
  const int wv    = tid >> 5;
  const int gbase = blockIdx.x * 128;
  const int mtile = blockIdx.x * 8 + wv;
  const bool active = (mtile < Mtiles);        // wave-uniform predicate

#if __has_builtin(__builtin_amdgcn_tensor_load_to_lds) && \
    __has_builtin(__builtin_amdgcn_s_wait_tensorcnt)
  if (wv == 0) {
    // ---- Tensor DMA descriptor (D#): 2-D bf16 tile [rows x 128] -> LDS ----
    const unsigned rows  = (unsigned)min(128, Nc - gbase);  // HW zero-fills OOB
    const unsigned long long ga =
        (unsigned long long)(const void*)(HCB + (size_t)gbase * DD);
    const unsigned ldsa = (unsigned)(unsigned long long)(const void*)sHC;

    u32x4 g0 = {0u, 0u, 0u, 0u};
    g0[0] = 1u;                                       // count=1, user mode
    g0[1] = ldsa;                                     // lds_addr
    g0[2] = (unsigned)(ga & 0xFFFFFFFFull);           // global_addr[31:0]
    g0[3] = (unsigned)((ga >> 32) & 0x1FFFFFFull)     // global_addr[56:32]
          | (2u << 30);                               // type = 2 ("image")

    i32x8 g1 = {0, 0, 0, 0, 0, 0, 0, 0};
    g1[0] = (1 << 16);                 // data_size = 1 (2 bytes/elt), mask=0
    g1[1] = (int)(128u << 16);         // tensor_dim0[15:0]=128 @bits[63:48]
    g1[2] = (int)((rows & 0xFFFFu) << 16); // dim0[31:16]=0, tensor_dim1[15:0]=rows
    g1[3] = (int)(128u << 16);         // tensor_dim1[31:16]=0, tile_dim0=128
    g1[4] = 128;                       // tile_dim1=128, tile_dim2=0
    g1[5] = 128;                       // tensor_dim0_stride[31:0]=128
    // g1[6], g1[7]: stride hi / dim1_stride = 0 (2-D tile)

    i32x4 gz = {0, 0, 0, 0};           // groups 2/3 unused (<=2D tensor)
#if defined(__clang_major__) && (__clang_major__ >= 23)
    i32x8 gz8 = {0, 0, 0, 0, 0, 0, 0, 0};
    __builtin_amdgcn_tensor_load_to_lds(g0, g1, gz, gz, gz8, 0);
#else
    __builtin_amdgcn_tensor_load_to_lds(g0, g1, gz, gz, 0);
#endif
    __builtin_amdgcn_s_wait_tensorcnt((short)0);      // s_wait_tensorcnt 0
  }
  __syncthreads();                                    // tile visible to all waves
#else
  // Fallback: cooperative staging of the bf16 tile.
  for (int idx = tid; idx < 128 * DD; idx += 256) {
    const int r = idx >> 7, col = idx & 127;
    const int gr = gbase + r;
    sHC[idx] = (gr < Nc) ? HCB[(size_t)gr * DD + col] : (__bf16)0.f;
  }
  __syncthreads();
#endif

  const int lrow16 = lane & 15;
  const int khalf  = (lane < 16) ? 0 : 16;
  const int mofs   = (lane < 16) ? 0 : 8;
  const int lm     = wv * 16;                   // local row base of this wave

  v8f acc2[8] = {};
  for (int hc0 = 0; hc0 < HH; hc0 += 128) {
    if (active) {
      v8f acc1[8] = {};
#pragma unroll
      for (int k0 = 0; k0 < DD; k0 += 32) {
        v16bf a = *(const v16bf*)(sHC + (lm + lrow16) * DD + k0 + khalf);
#pragma unroll
        for (int n = 0; n < 8; ++n) {
          v16bf b = *(const v16bf*)(W1B + (size_t)(hc0 + n * 16 + lrow16) * DD
                                        + k0 + khalf);
          acc1[n] = wmma_bf16(a, b, acc1[n]);
        }
      }
      // +b1, relu, stage as bf16 for GEMM2's A operand
#pragma unroll
      for (int n = 0; n < 8; ++n) {
        const float bias = B1[hc0 + n * 16 + lrow16];
#pragma unroll
        for (int r = 0; r < 8; ++r) {
          float v = acc1[n][r] + bias;
          v = (v > 0.f) ? v : 0.f;
          sT[(lm + r + mofs) * 128 + n * 16 + lrow16] = (__bf16)v;
        }
      }
    }
    __syncthreads();
    if (active) {
#pragma unroll
      for (int k0 = 0; k0 < 128; k0 += 32) {
        v16bf a = *(const v16bf*)(sT + (lm + lrow16) * 128 + k0 + khalf);
#pragma unroll
        for (int n = 0; n < 8; ++n) {
          v16bf b = *(const v16bf*)(W2B + (size_t)(n * 16 + lrow16) * HH
                                        + hc0 + k0 + khalf);
          acc2[n] = wmma_bf16(a, b, acc2[n]);
        }
      }
    }
    __syncthreads();
  }

  if (active) {
#pragma unroll
    for (int n = 0; n < 8; ++n) {
      const int col = n * 16 + lrow16;
      const float bias = B2[col];
#pragma unroll
      for (int r = 0; r < 8; ++r) {
        const int grow = gbase + lm + r + mofs;
        Y[(size_t)grow * DD + col] =
            acc2[n][r] + bias + HC[(size_t)grow * DD + col];   // pre-LN
      }
    }
  }
}

// ---------------------------------------------------------------------------
// Kernel 5: in-place LayerNorm over last dim (128). One block per row.
// ---------------------------------------------------------------------------
__global__ void k_ln(float* __restrict__ Y, const float* __restrict__ G,
                     const float* __restrict__ B) {
  const int row = blockIdx.x;
  const int tid = threadIdx.x;
  __shared__ float red[128];
  const float v = Y[(size_t)row * DD + tid];
  red[tid] = v;
  __syncthreads();
#pragma unroll
  for (int off = 64; off > 0; off >>= 1) {
    if (tid < off) red[tid] += red[tid + off];
    __syncthreads();
  }
  const float mu = red[0] * (1.f / 128.f);
  __syncthreads();
  const float d = v - mu;
  red[tid] = d * d;
  __syncthreads();
#pragma unroll
  for (int off = 64; off > 0; off >>= 1) {
    if (tid < off) red[tid] += red[tid + off];
    __syncthreads();
  }
  const float var = red[0] * (1.f / 128.f);
  Y[(size_t)row * DD + tid] = d * __frsqrt_rn(var + LN_EPS) * G[tid] + B[tid];
}

// ---------------------------------------------------------------------------
extern "C" void kernel_launch(void* const* d_in, const int* in_sizes, int n_in,
                              void* d_out, int out_size, void* d_ws, size_t ws_size,
                              hipStream_t stream) {
  const float* s      = (const float*)d_in[0];
  const float* c      = (const float*)d_in[1];
  const int*   src    = (const int*)d_in[2];
  const int*   dst    = (const int*)d_in[3];
  const float* Wfc    = (const float*)d_in[4];
  const float* a_attn = (const float*)d_in[5];
  const float* w1     = (const float*)d_in[6];
  const float* b1     = (const float*)d_in[7];
  const float* w2     = (const float*)d_in[8];
  const float* b2     = (const float*)d_in[9];
  const float* ln_g   = (const float*)d_in[10];
  const float* ln_b   = (const float*)d_in[11];

  const int Ns = in_sizes[0] / DD;   // 50000
  const int Nc = in_sizes[1] / DD;   // 50000
  const int E  = in_sizes[2];        // 1600000

  // Workspace carve-up (256B aligned segments):
  //   z [Ns,128] f32 | el [Ns] f32 | hc [Nc,128] f32 | hcb [Nc,128] bf16
  //   Wb [128,128] bf16 | w1b [512,128] bf16 | w2b [128,512] bf16
  char* ws = (char*)d_ws;
  auto takeb = [&](size_t bytes) {
    char* p = ws;
    ws += (bytes + 255) & ~(size_t)255;
    return p;
  };
  float*  z   = (float*) takeb((size_t)Ns * DD * 4);
  float*  el  = (float*) takeb((size_t)Ns * 4);
  float*  hc  = (float*) takeb((size_t)Nc * DD * 4);
  __bf16* hcb = (__bf16*)takeb((size_t)Nc * DD * 2);
  __bf16* Wb  = (__bf16*)takeb((size_t)DD * DD * 2);
  __bf16* w1b = (__bf16*)takeb((size_t)HH * DD * 2);
  __bf16* w2b = (__bf16*)takeb((size_t)DD * HH * 2);

  float* y = (float*)d_out;

  const int mtS = Ns / 16;   // 50000/16 = 3125 (exact)
  const int mtC = Nc / 16;

  // weight pre-conversion (tiny; removes cvt + halves weight bytes in GEMMs)
  k_cvt<<<(DD * DD + 255) / 256, 256, 0, stream>>>(Wfc, Wb, DD * DD);
  k_cvt<<<(HH * DD + 255) / 256, 256, 0, stream>>>(w1, w1b, HH * DD);
  k_cvt<<<(DD * HH + 255) / 256, 256, 0, stream>>>(w2, w2b, DD * HH);

  k_gemm_z<<<(mtS + 7) / 8, 256, 0, stream>>>(s, Wb, z, mtS);
  k_row_dot<<<(Ns + 7) / 8, 256, 0, stream>>>(z, a_attn, el, Ns);
  k_gat_agg<<<Nc, 128, 0, stream>>>(el, z, src, dst, c, hc, hcb, E);
  k_ffn<<<(mtC + 7) / 8, 256, 0, stream>>>(hc, hcb, w1b, b1, w2b, b2, y, Nc, mtC);
  k_ln<<<Nc, 128, 0, stream>>>(y, ln_g, ln_b);
}
